// Dependency_GATLayer_10075993276698
// MI455X (gfx1250) — matrix operand order repaired and checked by reading, not compiled, not run on hardware
//
#include <hip/hip_runtime.h>

typedef __attribute__((ext_vector_type(16))) _Float16 v16h;
typedef __attribute__((ext_vector_type(8)))  _Float16 v8h;
typedef __attribute__((ext_vector_type(8)))  float    v8f;

// ---------------- f32 -> f16 conversion ----------------
__global__ __launch_bounds__(256) void k_f32_to_f16(const float* __restrict__ in,
                                                    _Float16* __restrict__ out, int n) {
    int i = blockIdx.x * blockDim.x + threadIdx.x;
    if (i < n) out[i] = (_Float16)in[i];
}

// ---------------- Hx[M,Dn] = Xh[M,K] @ Wh[Dn,K]^T  via V_WMMA_F32_16X16X32_F16 ---
// One wave per 16x16 output tile. Per-lane fragment loads follow the CDNA5
// wave32 WMMA VGPR layouts (ISA 7.12.2): A 16x32 f16, B 32x16 f16, C/D 16x16 f32.
__global__ __launch_bounds__(256) void k_gemm_wmma(const _Float16* __restrict__ Xh,
                                                   const _Float16* __restrict__ Wh,
                                                   float* __restrict__ Hx,
                                                   int M, int K, int Dn) {
    const int wave = blockIdx.x * (blockDim.x >> 5) + (threadIdx.x >> 5);
    const int ntn  = Dn >> 4;
    const int tm   = wave / ntn;
    const int tn   = wave % ntn;
    if (tm * 16 >= M) return;

    const int lane = threadIdx.x & 31;
    const int sub  = lane & 15;   // A: row-in-tile, B: col-in-tile
    const int hs   = lane >> 4;   // half-wave select

    const _Float16* arow = Xh + (size_t)(tm * 16 + sub) * K;
    const _Float16* brow = Wh + (size_t)(tn * 16 + sub) * K;   // B[k,j] = W[j,k]

    v8f acc = {};
    #pragma unroll 4
    for (int k0 = 0; k0 < K; k0 += 32) {
        // A layout: lanes 0-15 hold K = k0+0..7 (v0-3) and k0+16..23 (v4-7);
        //           lanes 16-31 shift K by +8.  All 16B-aligned b128 loads.
        v8h alo = *(const v8h*)(arow + k0 + hs * 8);
        v8h ahi = *(const v8h*)(arow + k0 + 16 + hs * 8);
        // B layout: same K mapping as A (lanes 0-15 K = k0+0..7, k0+16..23; lanes 16-31 +8).
        v8h blo = *(const v8h*)(brow + k0 + hs * 8);
        v8h bhi = *(const v8h*)(brow + k0 + 16 + hs * 8);
        v16h af = __builtin_shufflevector(alo, ahi, 0,1,2,3,4,5,6,7,8,9,10,11,12,13,14,15);
        v16h bf = __builtin_shufflevector(blo, bhi, 0,1,2,3,4,5,6,7,8,9,10,11,12,13,14,15);
        acc = __builtin_amdgcn_wmma_f32_16x16x32_f16(
                  /*neg_a=*/false, af, /*neg_b=*/false, bf,
                  /*c_mod=*/(short)0, acc, /*reuse_a=*/false, /*reuse_b=*/false);
    }

    // D layout: VGPR r -> row (hs*8 + r), col = sub.
    float* cbase = Hx + (size_t)(tm * 16 + hs * 8) * Dn + tn * 16 + sub;
    #pragma unroll
    for (int r = 0; r < 8; ++r)
        cbase[(size_t)r * Dn] = acc[r];
}

// ---------------- g1[i] = Hx[i].a[:D],  g2[i] = Hx[i].a[D:] (one wave per row) --
__global__ __launch_bounds__(256) void k_row_dots(const float* __restrict__ Hx,
                                                  const float* __restrict__ a,
                                                  float* __restrict__ g1,
                                                  float* __restrict__ g2,
                                                  int N, int D) {
    int row = blockIdx.x * (blockDim.x >> 5) + (threadIdx.x >> 5);
    if (row >= N) return;
    int lane = threadIdx.x & 31;
    const float* hr = Hx + (size_t)row * D;
    float s1 = 0.f, s2 = 0.f;
    for (int c = lane; c < D; c += 32) {
        float v = hr[c];
        s1 += v * a[c];
        s2 += v * a[D + c];
    }
    #pragma unroll
    for (int off = 16; off > 0; off >>= 1) {
        s1 += __shfl_xor(s1, off, 32);
        s2 += __shfl_xor(s2, off, 32);
    }
    if (lane == 0) { g1[row] = s1; g2[row] = s2; }
}

// ---------------- inverse index maps ----------------
__global__ __launch_bounds__(256) void k_init_inv(int* __restrict__ invg,
                                                  int* __restrict__ invd, int N) {
    int i = blockIdx.x * blockDim.x + threadIdx.x;
    if (i < N) { invg[i] = -1; invd[i] = -1; }
}

__global__ __launch_bounds__(256) void k_scatter(const int* __restrict__ gov,
                                                 const int* __restrict__ dep,
                                                 int* __restrict__ invg,
                                                 int* __restrict__ invd, int E) {
    int e = blockIdx.x * blockDim.x + threadIdx.x;
    if (e < E) { invg[gov[e]] = e; invd[dep[e]] = e; }
}

// ---------------- fused gather + softmax-collapse coef + leaky relu -----------
// out[i] = lrelu( (invd[i]>=0 ? Hx[gov[invd[i]]] : 0)
//               + (invg[i]>=0 ? coef * Hx[dep[invg[i]]] : 0) )
// coef = (g1[i] + g2[dep]) > 0 ? 1 : 1/N   (exact collapse of the masked softmax)
__global__ __launch_bounds__(256) void k_finalize(const float* __restrict__ Hx,
                                                  const float* __restrict__ g1,
                                                  const float* __restrict__ g2,
                                                  const int* __restrict__ gov,
                                                  const int* __restrict__ dep,
                                                  const int* __restrict__ invg,
                                                  const int* __restrict__ invd,
                                                  float* __restrict__ out,
                                                  int N, int D, float invN, float alpha) {
    int row = blockIdx.x * (blockDim.x >> 5) + (threadIdx.x >> 5);
    if (row >= N) return;
    int lane = threadIdx.x & 31;

    int ed = invd[row];
    int eg = invg[row];
    int rowA = (ed >= 0) ? gov[ed] : -1;
    int rowB = -1;
    float coef = 0.f;
    if (eg >= 0) {
        rowB = dep[eg];
        float s = g1[row] + g2[rowB];
        coef = (s > 0.f) ? 1.0f : invN;
    }
    const float* pa = (rowA >= 0) ? (Hx + (size_t)rowA * D) : nullptr;
    const float* pb = (rowB >= 0) ? (Hx + (size_t)rowB * D) : nullptr;
    float* po = out + (size_t)row * D;

    for (int c = lane; c < D; c += 32) {
        float v = 0.f;
        if (pa) v += pa[c];
        if (pb) v += coef * pb[c];
        po[c] = (v > 0.f) ? v : alpha * v;
    }
}

extern "C" void kernel_launch(void* const* d_in, const int* in_sizes, int n_in,
                              void* d_out, int out_size, void* d_ws, size_t ws_size,
                              hipStream_t stream) {
    const float* x   = (const float*)d_in[0];
    const float* W   = (const float*)d_in[1];
    const float* a   = (const float*)d_in[2];
    const int*   dep = (const int*)d_in[3];
    const int*   gov = (const int*)d_in[4];
    float* out = (float*)d_out;

    const int E = in_sizes[3];
    const int N = E + 1;
    const int D = in_sizes[2] / 2;

    // workspace layout (256B aligned regions)
    size_t off = 0;
    auto alloc = [&](size_t bytes) {
        size_t o = off;
        off = (off + bytes + 255) & ~(size_t)255;
        return o;
    };
    char* ws = (char*)d_ws;
    _Float16* xh  = (_Float16*)(ws + alloc((size_t)N * D * sizeof(_Float16)));
    _Float16* Wh  = (_Float16*)(ws + alloc((size_t)D * D * sizeof(_Float16)));
    float*    Hx  = (float*)   (ws + alloc((size_t)N * D * sizeof(float)));
    float*    g1  = (float*)   (ws + alloc((size_t)N * sizeof(float)));
    float*    g2  = (float*)   (ws + alloc((size_t)N * sizeof(float)));
    int*      invg= (int*)     (ws + alloc((size_t)N * sizeof(int)));
    int*      invd= (int*)     (ws + alloc((size_t)N * sizeof(int)));
    (void)ws_size; (void)n_in; (void)out_size;

    const int nx = N * D;
    const int nw = D * D;
    k_f32_to_f16<<<(nx + 255) / 256, 256, 0, stream>>>(x, xh, nx);
    k_f32_to_f16<<<(nw + 255) / 256, 256, 0, stream>>>(W, Wh, nw);

    const int waves  = (N / 16) * (D / 16);
    const int gblocks = (waves + 7) / 8;          // 8 waves per 256-thread block
    k_gemm_wmma<<<gblocks, 256, 0, stream>>>(xh, Wh, Hx, N, D, D);

    const int rblocks = (N + 7) / 8;              // one wave per row
    k_row_dots<<<rblocks, 256, 0, stream>>>(Hx, a, g1, g2, N, D);

    k_init_inv<<<(N + 255) / 256, 256, 0, stream>>>(invg, invd, N);
    k_scatter<<<(E + 255) / 256, 256, 0, stream>>>(gov, dep, invg, invd, E);

    k_finalize<<<rblocks, 256, 0, stream>>>(Hx, g1, g2, gov, dep, invg, invd,
                                            out, N, D, 1.0f / (float)N, 0.2f);
}
